// FeaturePropagation_65283502899645
// MI455X (gfx1250) — matrix operand order repaired and checked by reading, not compiled, not run on hardware
//
#include <hip/hip_runtime.h>
#include <hip/hip_bf16.h>
#include <math.h>

// ---------------------------------------------------------------- constants
constexpr int BATCH = 4;
constexpr int M_FINE = 32768;
constexpr int N_COARSE = 2048;
constexpr int CH = 256;          // C_IN == C_OUT
constexpr float EPS = 1e-8f;

// ---------------------------------------------------------------- types
typedef __attribute__((ext_vector_type(16))) __bf16 v16bf;
typedef __attribute__((ext_vector_type(8)))  float  v8f;
typedef __attribute__((ext_vector_type(4)))  float  fv4;
typedef __attribute__((ext_vector_type(4)))  unsigned uv4;

union Frag {            // one wave-lane's share of a 16x32 / 32x16 bf16 matrix
    uv4      q[2];      // two 16-byte chunks
    unsigned u[8];
    v16bf    v;
};

__device__ inline unsigned umn(unsigned a, unsigned b) { return a < b ? a : b; }
__device__ inline unsigned umx(unsigned a, unsigned b) { return a > b ? a : b; }

// round-to-nearest-even f32 -> bf16 (bits)
__device__ inline unsigned short f2bf(float x) {
    unsigned u = __float_as_uint(x);
    u += 0x7FFFu + ((u >> 16) & 1u);
    return (unsigned short)(u >> 16);
}
__device__ inline unsigned pack_bf16(float lo, float hi) {
    return (unsigned)f2bf(lo) | ((unsigned)f2bf(hi) << 16);
}

// =====================================================================
// Kernel 1: W1/W2 (f32 [k][n]) -> transposed bf16 Wt[n][k] so that the
// per-lane WMMA B fragment (two runs of 8 consecutive K of one column n) is two
// contiguous 16-byte segments. Writes coalesced; strided reads stay in L2.
// =====================================================================
__global__ void convert_weights_kernel(const float* __restrict__ W1,
                                       const float* __restrict__ W2,
                                       unsigned short* __restrict__ W1t,
                                       unsigned short* __restrict__ W2t) {
    int id = blockIdx.x * 256 + threadIdx.x;          // 0 .. 2*256*256-1
    const float* W = (id < CH * CH) ? W1 : W2;
    unsigned short* Wt = (id < CH * CH) ? W1t : W2t;
    int e = id & (CH * CH - 1);
    int n = e >> 8;
    int k = e & 255;                                  // consecutive tid -> consecutive k
    Wt[n * CH + k] = f2bf(W[k * CH + n]);             // coalesced write
}

// =====================================================================
// Kernel 2: KNN(3) + inverse-distance interpolation -> bf16 interp.
// One block = 256 fine points of one batch. Coarse coords + |c|^2 live in
// LDS; top-3 kept as packed (d2bits | idx) u32 keys via a min/max network
// (f32 bit pattern is monotone in u32 for non-negative values).
// =====================================================================
__global__ void __launch_bounds__(256) knn_interp_kernel(
        const float* __restrict__ fine,
        const float* __restrict__ coarse,
        const float* __restrict__ feats,
        unsigned short* __restrict__ interp) {
    __shared__ fv4   cc[N_COARSE];     // {x,y,z,|c|^2}  32 KB
    __shared__ float sw[256][4];       // normalized weights
    __shared__ int   si[256][4];       // knn indices

    const int tid   = threadIdx.x;
    const int b     = blockIdx.x / (M_FINE / 256);
    const int chunk = blockIdx.x % (M_FINE / 256);

    // stage coarse points
    const float* cb = coarse + (size_t)b * N_COARSE * 3;
    for (int j = tid; j < N_COARSE; j += 256) {
        float x = cb[3 * j], y = cb[3 * j + 1], z = cb[3 * j + 2];
        fv4 v; v.x = x; v.y = y; v.z = z; v.w = x * x + y * y + z * z;
        cc[j] = v;
    }
    __syncthreads();

    // per-thread scan of all coarse points: d^2 = |f|^2 - 2 f.c + |c|^2
    const int m = chunk * 256 + tid;
    const float* fp = fine + ((size_t)b * M_FINE + m) * 3;
    const float fx = fp[0], fy = fp[1], fz = fp[2];
    const float f2 = fx * fx + fy * fy + fz * fz;
    const float nx = -2.0f * fx, ny = -2.0f * fy, nz = -2.0f * fz;

    unsigned k0 = ~0u, k1 = ~0u, k2 = ~0u;
#pragma unroll 4
    for (int j = 0; j < N_COARSE; ++j) {
        fv4 c = cc[j];
        float s = fmaf(nx, c.x, fmaf(ny, c.y, fmaf(nz, c.z, c.w + f2))); // d^2
        s = fmaxf(s, 0.0f);                       // keep f32 bits monotone as u32
        unsigned key = (__float_as_uint(s) & 0xFFFFF800u) | (unsigned)j;
        unsigned lo = umn(key, k0), hi = umx(key, k0); k0 = lo;   // sorted insert
        unsigned lo1 = umn(hi, k1), hi1 = umx(hi, k1); k1 = lo1;
        k2 = umn(hi1, k2);
    }

    unsigned kk[3] = {k0, k1, k2};
    float w[3]; int id3[3]; float wsum = 0.0f;
#pragma unroll
    for (int t = 0; t < 3; ++t) {
        id3[t] = (int)(kk[t] & 0x7FFu);
        float d = sqrtf(__uint_as_float(kk[t] & 0xFFFFF800u));
        w[t] = 1.0f / (d + EPS);
        wsum += w[t];
    }
#pragma unroll
    for (int t = 0; t < 3; ++t) { sw[tid][t] = w[t] / wsum; si[tid][t] = id3[t]; }
    __syncthreads();

    // cooperative, coalesced gather: each thread owns one channel
    const float* fb = feats + (size_t)b * N_COARSE * CH;
    unsigned short* ob = interp + ((size_t)b * M_FINE + chunk * 256) * CH;
    for (int r = 0; r < 256; ++r) {
        int   i0 = si[r][0], i1 = si[r][1], i2 = si[r][2];
        float w0 = sw[r][0], w1 = sw[r][1], w2 = sw[r][2];
        float acc = w0 * fb[(size_t)i0 * CH + tid]
                  + w1 * fb[(size_t)i1 * CH + tid]
                  + w2 * fb[(size_t)i2 * CH + tid];
        ob[(size_t)r * CH + tid] = f2bf(acc);
    }
}

// =====================================================================
// Kernel 3: fused  out = gelu(interp@W1+b1) @ W2 + b2   with bf16 WMMA.
// Block = 256 threads = 8 waves = 4 M-tiles(16 rows) x 2 N-halves(128 cols).
// h (64x256 f32, 64KB) is exchanged through LDS between the two GEMMs.
// A sched_barrier(0) separates the load group from the WMMA burst of each
// K-step so the backend cannot sink loads next to their consumers: all 18 (GEMM1)
// b128 loads issue as one clause and each WMMA drains with a partial
// s_wait_loadcnt while later loads stay in flight (covered by the ~7
// preceding multi-cycle XDL WMMAs).
// Fragment layouts follow CDNA5 ISA 7.12.2:
//   A 16x32 bf16 : lane l -> row l%16, two contiguous 16B K-segments
//   B 32x16 bf16 : lane l -> col l%16, two contiguous 16B K-segments (of Wt)
//   C/D 16x16 f32: vgpr i -> row i + 8*(l/16), col l%16
// =====================================================================
__global__ void __launch_bounds__(256, 2) fused_mlp_kernel(
        const unsigned short* __restrict__ interp,
        const unsigned short* __restrict__ W1t,
        const unsigned short* __restrict__ W2t,
        const float* __restrict__ b1,
        const float* __restrict__ b2,
        float* __restrict__ out) {
    __shared__ float hbuf[64 * CH];                   // 64 KB

    const int tid  = threadIdx.x;
    const int wave = tid >> 5, lane = tid & 31;
    const int mt = wave >> 1;                          // M tile (0..3)
    const int nh = wave & 1;                           // N half (0..1)
    const int g  = lane >> 4, ln = lane & 15;
    const size_t rowblk = (size_t)blockIdx.x * 64;

    const unsigned short* bbase1 = W1t + (size_t)(nh * 128 + ln) * CH;
    const unsigned short* bbase2 = W2t + (size_t)(nh * 128 + ln) * CH;

    // warm this wave's W2t fragments into the cache while GEMM1 runs
#pragma unroll
    for (int nt = 0; nt < 8; ++nt)
        __builtin_prefetch(bbase2 + (size_t)nt * 16 * CH + g * 16, 0, 3);

    // ---------------- GEMM1: interp(bf16) @ W1t --------------------
    v8f acc[8] = {};
    const unsigned short* arow = interp + (rowblk + mt * 16 + ln) * CH;
#pragma unroll
    for (int kt = 0; kt < 8; ++kt) {
        Frag fa;
        const unsigned short* ap = arow + kt * 32 + g * 8;
        fa.q[0] = *(const uv4*)(ap);                   // K = kb .. kb+7
        fa.q[1] = *(const uv4*)(ap + 16);              // K = kb+16 .. kb+23
        Frag fb[8];
        const unsigned short* bp = bbase1 + kt * 32 + g * 8;
#pragma unroll
        for (int nt = 0; nt < 8; ++nt) {               // one big load clause
            const unsigned short* p = bp + (size_t)nt * 16 * CH;
            fb[nt].q[0] = *(const uv4*)(p);
            fb[nt].q[1] = *(const uv4*)(p + 16);
        }
        __builtin_amdgcn_sched_barrier(0);             // keep loads above WMMAs
#pragma unroll
        for (int nt = 0; nt < 8; ++nt)                 // WMMA burst
            acc[nt] = __builtin_amdgcn_wmma_f32_16x16x32_bf16(
                false, fa.v, false, fb[nt].v, (short)0, acc[nt], false, false);
    }

    // bias + exact GELU -> LDS (f32, conflict-free: lanes write adjacent cols)
#pragma unroll
    for (int nt = 0; nt < 8; ++nt) {
        const int n = nh * 128 + nt * 16 + ln;
        const float bias = b1[n];
#pragma unroll
        for (int i = 0; i < 8; ++i) {
            float x = acc[nt][i] + bias;
            hbuf[(mt * 16 + 8 * g + i) * CH + n] =
                0.5f * x * (1.0f + erff(x * 0.70710678118654752f));
        }
    }
    __syncthreads();

    // ---------------- GEMM2: h(bf16 on the fly) @ W2t --------------
    v8f acc2[8] = {};
#pragma unroll
    for (int kt = 0; kt < 8; ++kt) {
        Frag fb[8];
        const unsigned short* bp = bbase2 + kt * 32 + g * 8;
#pragma unroll
        for (int nt = 0; nt < 8; ++nt) {               // one big load clause
            const unsigned short* p = bp + (size_t)nt * 16 * CH;
            fb[nt].q[0] = *(const uv4*)(p);
            fb[nt].q[1] = *(const uv4*)(p + 16);
        }
        const float* hp = hbuf + (mt * 16 + ln) * CH + kt * 32 + g * 8;
        fv4 h0 = *(const fv4*)(hp);
        fv4 h1 = *(const fv4*)(hp + 4);
        fv4 h2 = *(const fv4*)(hp + 16);
        fv4 h3 = *(const fv4*)(hp + 20);
        Frag fa;
        fa.u[0] = pack_bf16(h0.x, h0.y); fa.u[1] = pack_bf16(h0.z, h0.w);
        fa.u[2] = pack_bf16(h1.x, h1.y); fa.u[3] = pack_bf16(h1.z, h1.w);
        fa.u[4] = pack_bf16(h2.x, h2.y); fa.u[5] = pack_bf16(h2.z, h2.w);
        fa.u[6] = pack_bf16(h3.x, h3.y); fa.u[7] = pack_bf16(h3.z, h3.w);
        __builtin_amdgcn_sched_barrier(0);             // keep loads above WMMAs
#pragma unroll
        for (int nt = 0; nt < 8; ++nt)                 // WMMA burst
            acc2[nt] = __builtin_amdgcn_wmma_f32_16x16x32_bf16(
                false, fa.v, false, fb[nt].v, (short)0, acc2[nt], false, false);
    }

    // bias2 + store f32 output
#pragma unroll
    for (int nt = 0; nt < 8; ++nt) {
        const int n = nh * 128 + nt * 16 + ln;
        const float bias = b2[n];
        float* orow = out + (rowblk + mt * 16 + 8 * g) * CH + n;
#pragma unroll
        for (int i = 0; i < 8; ++i)
            orow[(size_t)i * CH] = acc2[nt][i] + bias;
    }
}

// =====================================================================
extern "C" void kernel_launch(void* const* d_in, const int* in_sizes, int n_in,
                              void* d_out, int out_size, void* d_ws, size_t ws_size,
                              hipStream_t stream) {
    (void)in_sizes; (void)n_in; (void)out_size; (void)ws_size;
    const float* fine   = (const float*)d_in[0];
    const float* coarse = (const float*)d_in[1];
    const float* feats  = (const float*)d_in[2];
    const float* W1     = (const float*)d_in[3];
    const float* b1     = (const float*)d_in[4];
    const float* W2     = (const float*)d_in[5];
    const float* b2     = (const float*)d_in[6];
    float* out = (float*)d_out;

    // workspace layout: [ interp bf16 : 64 MB ][ W1t bf16 : 128 KB ][ W2t bf16 : 128 KB ]
    const size_t INTERP_BYTES = (size_t)BATCH * M_FINE * CH * sizeof(unsigned short);
    unsigned short* interp = (unsigned short*)d_ws;
    unsigned short* W1t = (unsigned short*)((char*)d_ws + INTERP_BYTES);
    unsigned short* W2t = W1t + CH * CH;

    convert_weights_kernel<<<dim3((2 * CH * CH) / 256), dim3(256), 0, stream>>>(
        W1, W2, W1t, W2t);
    knn_interp_kernel<<<dim3(BATCH * (M_FINE / 256)), dim3(256), 0, stream>>>(
        fine, coarse, feats, interp);
    fused_mlp_kernel<<<dim3((BATCH * M_FINE) / 64), dim3(256), 0, stream>>>(
        interp, W1t, W2t, b1, b2, out);
}
